// Synchronized_fuzzy_Convolution_layer_75496935129477
// MI455X (gfx1250) — matrix operand order repaired and checked by reading, not compiled, not run on hardware
//
#include <hip/hip_runtime.h>

// ---------------------------------------------------------------------------
// out[n] = ReLU( diag(d) (S*F) diag(d) @ H @ theta1 ),  d = rowsum(A)^-0.5
// ROW=70 padded to M/N=80 (5 tiles of 16), K=96 (3 steps of 32) for
// v_wmma_f32_16x16x32_f16. f16 operands, fp32 accumulation.
// ---------------------------------------------------------------------------

typedef _Float16 v16h __attribute__((ext_vector_type(16)));
typedef _Float16 v8h  __attribute__((ext_vector_type(8)));
typedef float    v8f  __attribute__((ext_vector_type(8)));
typedef unsigned int v4u __attribute__((ext_vector_type(4)));

#define ROW   70
#define RR    (ROW * ROW)      // 4900
#define LDW   104              // padded row stride in f16 elems (avoids bank conflicts)
#define MP    80               // padded M/N
#define KP    96               // padded K (multiple of 32)
#define TILE16S (MP * LDW / 8) // # of 16B chunks per LDS tile = 1040

union Frag { v16h v; v8h h[2]; };

__global__ __launch_bounds__(256)
void sfc_fused_kernel(const float* __restrict__ S,
                      const float* __restrict__ F,
                      const float* __restrict__ H,
                      const float* __restrict__ A,
                      const float* __restrict__ Th,
                      float* __restrict__ Out)
{
    __shared__ __align__(16) _Float16 sSF[MP * LDW]; // A-operand GEMM1: [M][K]
    __shared__ __align__(16) _Float16 sHt[MP * LDW]; // B-operand GEMM1: [N][K] (H^T)
    __shared__ __align__(16) _Float16 sT [MP * LDW]; // A-operand GEMM2: [M][K]
    __shared__ __align__(16) _Float16 sTh[MP * LDW]; // B-operand GEMM2: [N][K] (theta^T)
    __shared__ float sd[ROW];

    const int tid  = threadIdx.x;
    const int lane = tid & 31;
    const int wv   = tid >> 5;
    const int n    = blockIdx.x;
    const size_t base = (size_t)n * RR;

    // ---- Phase 0: zero-fill padded LDS tiles (ds_store_b128) ----
    {
        v4u z = {};
        v4u* p0 = (v4u*)sSF; v4u* p1 = (v4u*)sHt;
        v4u* p2 = (v4u*)sT;  v4u* p3 = (v4u*)sTh;
        for (int i = tid; i < TILE16S; i += 256) {
            p0[i] = z; p1[i] = z; p2[i] = z; p3[i] = z;
        }
    }

    // ---- Phase 1: d[r] = rsqrt(sum_c A[n][r][c]); coalesced + wave32 reduce ----
    for (int r = wv; r < ROW; r += 8) {
        float s = 0.f;
        for (int c = lane; c < ROW; c += 32) s += A[base + r * ROW + c];
        #pragma unroll
        for (int m = 16; m > 0; m >>= 1) s += __shfl_xor(s, m, 32);
        if (lane == 0) sd[r] = rsqrtf(s);
    }
    __syncthreads();

    // ---- Phase 2: build f16 operands in LDS ----
    for (int idx = tid; idx < RR; idx += 256) {
        int i = idx / ROW;
        int j = idx - i * ROW;
        float sf = sd[i] * sd[j] * S[base + idx] * F[base + idx];
        sSF[i * LDW + j] = (_Float16)sf;          // [M=i][K=j]
        sHt[j * LDW + i] = (_Float16)H[base + idx]; // H^T: [N=j][K=i]
        sTh[j * LDW + i] = (_Float16)Th[idx];       // theta^T: [N=j][K=i]
    }
    __syncthreads();

    // Per-lane fragment geometry (ISA 7.12.2, wave32):
    //  A: lanes 0-15 -> M=lane,    K chunks {kb+0..7, kb+16..23}
    //     lanes16-31 -> M=lane-16, K chunks {kb+8..15, kb+24..31}
    //  B: lanes 0-15 -> N=lane,    K chunks {kb+0..7, kb+16..23}
    //     lanes16-31 -> N=lane-16, K chunks {kb+8..15, kb+24..31}
    const int lm   = lane & 15;
    const int asel = (lane < 16) ? 0 : 8;
    const int bsel = (lane < 16) ? 0 : 16;
    const int mrow = ((lane < 16) ? 0 : 8); // C/D: vgpr r -> M = mt*16 + mrow + r

    // ---- Phase 3: GEMM1  T = SF @ H ----
    for (int t = wv; t < 25; t += 8) {
        const int mt = t / 5, nt = t - mt * 5;
        const _Float16* arow = &sSF[(mt * 16 + lm) * LDW];
        const _Float16* brow = &sHt[(nt * 16 + lm) * LDW];
        v8f acc = {};
        #pragma unroll
        for (int kb = 0; kb < KP; kb += 32) {
            Frag a, b;
            a.h[0] = *(const v8h*)(arow + kb + asel);
            a.h[1] = *(const v8h*)(arow + kb + 16 + asel);
            b.h[0] = *(const v8h*)(brow + kb + asel);
            b.h[1] = *(const v8h*)(brow + kb + 16 + asel);
            acc = __builtin_amdgcn_wmma_f32_16x16x32_f16(
                false, a.v, false, b.v, (short)0, acc, false, false);
        }
        const int m0 = mt * 16 + mrow;
        const int nc = nt * 16 + lm;
        #pragma unroll
        for (int r = 0; r < 8; ++r)
            sT[(m0 + r) * LDW + nc] = (_Float16)acc[r];
    }
    __syncthreads();

    // ---- Phase 4: GEMM2  Out = ReLU(T @ theta1) ----
    for (int t = wv; t < 25; t += 8) {
        const int mt = t / 5, nt = t - mt * 5;
        const _Float16* arow = &sT [(mt * 16 + lm) * LDW];
        const _Float16* brow = &sTh[(nt * 16 + lm) * LDW];
        v8f acc = {};
        #pragma unroll
        for (int kb = 0; kb < KP; kb += 32) {
            Frag a, b;
            a.h[0] = *(const v8h*)(arow + kb + asel);
            a.h[1] = *(const v8h*)(arow + kb + 16 + asel);
            b.h[0] = *(const v8h*)(brow + kb + asel);
            b.h[1] = *(const v8h*)(brow + kb + 16 + asel);
            acc = __builtin_amdgcn_wmma_f32_16x16x32_f16(
                false, a.v, false, b.v, (short)0, acc, false, false);
        }
        const int m0 = mt * 16 + mrow;
        const int nc = nt * 16 + lm;
        if (nc < ROW) {
            #pragma unroll
            for (int r = 0; r < 8; ++r) {
                const int mr = m0 + r;
                if (mr < ROW)
                    Out[base + mr * ROW + nc] = fmaxf(acc[r], 0.0f);
            }
        }
    }
}

extern "C" void kernel_launch(void* const* d_in, const int* in_sizes, int n_in,
                              void* d_out, int out_size, void* d_ws, size_t ws_size,
                              hipStream_t stream) {
    (void)n_in; (void)out_size; (void)d_ws; (void)ws_size;
    const float* S  = (const float*)d_in[0];
    const float* F  = (const float*)d_in[1];
    const float* H  = (const float*)d_in[2];
    const float* A  = (const float*)d_in[3];
    const float* Th = (const float*)d_in[4];
    float* Out = (float*)d_out;
    const int num = in_sizes[0] / RR; // 8192
    sfc_fused_kernel<<<num, 256, 0, stream>>>(S, F, H, A, Th, Out);
}
